// KMeans_53798760349790
// MI455X (gfx1250) — compile-verified, matrix operand order checked by reading, not run on hardware
//
#include <hip/hip_runtime.h>

// VQ / k-means codebook lookup for MI455X (gfx1250, wave32, WMMA bf16x3,
// async global->LDS double-buffered streaming).
// B=16, T=1024, D=256, K=8192.  N = 16384 rows.
//
// score(n,k) = ||e_k||^2 - 2 * x_n . e_k     (||x_n||^2 constant per row -> dropped)
// x.e via bf16 split-GEMM: x=xh+xl, e=eh+el,
//   x.e ~= xh.eh + xh.el + xl.eh   (fp32 accumulate; xl.el ~2^-18 rel, dropped)
// argmin fused into the GEMM epilogue; distances never materialized.
// emb bf16 planes streamed to LDS with GLOBAL_LOAD_ASYNC_TO_LDS_B128 (ASYNCcnt),
// double-buffered so the DMA for tile it+1 hides behind the 48 WMMAs of tile it.
// Buffer selection is integer-offset based (not pointer-array based) so all
// fragment reads keep addrspace(3) inference and lower to ds_load_b128.

typedef __attribute__((ext_vector_type(8)))  float  v8f;
typedef __attribute__((ext_vector_type(8)))  __bf16 v8bf;
typedef __attribute__((ext_vector_type(16))) __bf16 v16bf;

#define N_ROWS   16384
#define DIM      256
#define KCODES   8192
#define DPH      264            // LDS row stride in halfs: 528B == 4 dwords mod 64 banks
#define MTILE    32             // rows per block
#define CTILE    64             // codes per block-iteration
#define NITERS   (KCODES / CTILE)
#define XPLANE   (MTILE * DPH)  // halfs
#define EPLANE   (CTILE * DPH)  // halfs

static __device__ __forceinline__ unsigned short f32_to_bf16_rne(float f) {
    unsigned int u = __float_as_uint(f);
    u += 0x7FFFu + ((u >> 16) & 1u);          // round to nearest even
    return (unsigned short)(u >> 16);
}
static __device__ __forceinline__ float bf16_to_f32(unsigned short h) {
    return __uint_as_float(((unsigned int)h) << 16);
}
static __device__ __forceinline__ void split_bf16(float f, unsigned short& h, unsigned short& l) {
    h = f32_to_bf16_rne(f);
    l = f32_to_bf16_rne(f - bf16_to_f32(h));
}
static __device__ __forceinline__ v16bf ldfrag(const unsigned short* p) {
    v8bf c0 = *(const v8bf*)(p);
    v8bf c1 = *(const v8bf*)(p + 16);
    return __builtin_shufflevector(c0, c1, 0,1,2,3,4,5,6,7,8,9,10,11,12,13,14,15);
}
#define WMMA_BF16(A, B, C) \
    __builtin_amdgcn_wmma_f32_16x16x32_bf16(false, (A), false, (B), (short)0, (C), false, false)

// Low 32 bits of a generic LDS-aperture address == LDS byte offset (flat addressing
// truncates to addr[31:0] for the LDS segment).
static __device__ __forceinline__ unsigned lds_off(const void* p) {
    return (unsigned)(unsigned long long)p;
}
// GLOBAL_LOAD_ASYNC_TO_LDS_B128, GVS mode: mem = SADDR + VADDR(u32); LDS = VDST vgpr.
static __device__ __forceinline__ void async_copy_b128(unsigned ldsByteOff,
                                                       unsigned gByteOff,
                                                       unsigned long long sbase) {
    asm volatile("global_load_async_to_lds_b128 %0, %1, %2"
                 :: "v"(ldsByteOff), "v"(gByteOff), "s"(sbase)
                 : "memory");
}
static __device__ __forceinline__ void wait_async0() {
    asm volatile("s_wait_asynccnt 0x0" ::: "memory");
}

// ---------------------------------------------------- one-time: split emb + ||e||^2
__global__ __launch_bounds__(256) void prep_kernel(const float* __restrict__ emb,
                                                   unsigned short* __restrict__ e_hi,
                                                   unsigned short* __restrict__ e_lo,
                                                   float* __restrict__ enorm2) {
    const int lane = threadIdx.x & 31;
    const int code = blockIdx.x * 8 + (threadIdx.x >> 5);
    const float* row = emb + (size_t)code * DIM;
    float s = 0.0f;
    #pragma unroll
    for (int i = lane; i < DIM; i += 32) {
        const float f = row[i];
        s += f * f;
        unsigned short h, l;
        split_bf16(f, h, l);
        e_hi[(size_t)code * DIM + i] = h;
        e_lo[(size_t)code * DIM + i] = l;
    }
    #pragma unroll
    for (int m = 1; m < 32; m <<= 1) s += __shfl_xor(s, m, 32);
    if (lane == 0) enorm2[code] = s;
}

// ------------------------------------------------- fused bf16x3 GEMM + argmin + gather
__global__ __launch_bounds__(128) void vq_argmin_kernel(const float* __restrict__ x,
                                                        const float* __restrict__ emb,
                                                        const unsigned short* __restrict__ e_hi_g,
                                                        const unsigned short* __restrict__ e_lo_g,
                                                        const float* __restrict__ enorm2,
                                                        float* __restrict__ outq,
                                                        int* __restrict__ outidx) {
    extern __shared__ unsigned short smemu[];
    unsigned short* xs_hi = smemu;                     // XPLANE
    unsigned short* xs_lo = xs_hi + XPLANE;            // XPLANE
    unsigned short* es0   = xs_lo + XPLANE;            // 4 * EPLANE: hi0 | lo0 | hi1 | lo1
    float* sMinV = (float*)(es0 + 4 * EPLANE);         // 2 * MTILE
    int*   sMinI = (int*)(sMinV + 2 * MTILE);          // 2 * MTILE
    int*   sIdx  = sMinI + 2 * MTILE;                  // MTILE

    const int tid  = threadIdx.x;
    const int lane = tid & 31;
    const int wave = tid >> 5;
    const int rg   = wave >> 1;             // row group  (0/1): rows rg*16..+15
    const int cg   = wave & 1;              // code group (0/1): codes cg*32..+31 of tile
    const int rowBase = blockIdx.x * MTILE;

    const unsigned es0Off = lds_off(es0);   // LDS byte offset of plane block

    // per-lane staging geometry: chunk i = tid + 128j -> row = wave + 4j, col = lane*8
    const unsigned gStep    = 4u * DIM * 2u;    // 4 rows of global plane, bytes
    const unsigned lStep    = 4u * DPH * 2u;    // 4 rows of LDS plane, bytes
    const unsigned gOffBase = (unsigned)((wave * DIM + lane * 8) * 2);
    const unsigned lOffBase = (unsigned)((wave * DPH + lane * 8) * 2);

    // issue the async fill of emb tile (64 codes x 256 x {hi,lo}) into buffer b
    auto stage_async = [&](int b, int tileCode) {
        const unsigned long long sHi =
            (unsigned long long)(e_hi_g + (size_t)tileCode * DIM);
        const unsigned long long sLo =
            (unsigned long long)(e_lo_g + (size_t)tileCode * DIM);
        unsigned g  = gOffBase;
        unsigned lh = es0Off + (unsigned)(2 * EPLANE * sizeof(unsigned short)) * b + lOffBase;
        unsigned ll = lh + (unsigned)(EPLANE * sizeof(unsigned short));
        #pragma unroll
        for (int j = 0; j < 16; ++j) {      // 32 async b128 / wave / iter (<= 63 ASYNCcnt)
            async_copy_b128(lh, g, sHi);
            async_copy_b128(ll, g, sLo);
            g += gStep; lh += lStep; ll += lStep;
        }
    };

    // ---- stage + split x tile (32 rows x 256) once; reused for all 8192 codes
    for (int i = tid; i < MTILE * (DIM / 4); i += 128) {
        const int r = i >> 6;
        const int c = (i & 63) << 2;
        const float4 v = *(const float4*)(x + (size_t)(rowBase + r) * DIM + c);
        ushort4 h, l;
        split_bf16(v.x, h.x, l.x);
        split_bf16(v.y, h.y, l.y);
        split_bf16(v.z, h.z, l.z);
        split_bf16(v.w, h.w, l.w);
        *(ushort4*)(xs_hi + r * DPH + c) = h;
        *(ushort4*)(xs_lo + r * DPH + c) = l;
    }

    // prime buffer 0
    stage_async(0, 0);
    wait_async0();
    __syncthreads();

    // bf16 WMMA 16x16x32 fragment addressing (wave32):
    //  A: lane L -> row m = L&15, depths {kb..kb+7} u {kb+16..kb+23}, kb = 8*(L>>4)
    //  B: lane L -> col n = L&15 (code), same depth pattern
    //  C: vgpr v, lane L -> m = v + 8*(L>>4), n = L&15
    const int fi   = lane & 15;
    const int off8 = (lane >> 4) << 3;
    const unsigned short* aH = xs_hi + (rg * 16 + fi) * DPH + off8;
    const unsigned short* aL = xs_lo + (rg * 16 + fi) * DPH + off8;
    const int bofs0 = (cg * 32 + fi) * DPH + off8;      // within hi plane
    const int bofs1 = bofs0 + 16 * DPH;

    float bestV[8];
    int   bestI[8];
    #pragma unroll
    for (int v = 0; v < 8; ++v) { bestV[v] = 3.4e38f; bestI[v] = 0; }

    for (int it = 0; it < NITERS; ++it) {
        const int buf = it & 1;
        const int tileCode = it * CTILE;

        // kick off the DMA for the next tile into the other buffer (hidden by WMMAs)
        if (it + 1 < NITERS) stage_async(buf ^ 1, tileCode + CTILE);

        // integer-offset buffer select: keeps LDS addrspace -> ds_load_b128
        const unsigned short* eb = es0 + 2 * EPLANE * buf;
        const unsigned short* bH0 = eb + bofs0;
        const unsigned short* bL0 = eb + EPLANE + bofs0;
        const unsigned short* bH1 = eb + bofs1;
        const unsigned short* bL1 = eb + EPLANE + bofs1;

        v8f acc0 = {0.f,0.f,0.f,0.f,0.f,0.f,0.f,0.f};
        v8f acc1 = {0.f,0.f,0.f,0.f,0.f,0.f,0.f,0.f};
        #pragma unroll
        for (int d = 0; d < 8; ++d) {       // 8 depth steps of 32
            const int kb = d * 32;
            const v16bf ah = ldfrag(aH + kb);
            const v16bf al = ldfrag(aL + kb);
            {   // code sub-tile 0
                const v16bf bh = ldfrag(bH0 + kb);
                const v16bf bl = ldfrag(bL0 + kb);
                acc0 = WMMA_BF16(ah, bh, acc0);
                acc0 = WMMA_BF16(ah, bl, acc0);
                acc0 = WMMA_BF16(al, bh, acc0);
            }
            {   // code sub-tile 1
                const v16bf bh = ldfrag(bH1 + kb);
                const v16bf bl = ldfrag(bL1 + kb);
                acc1 = WMMA_BF16(ah, bh, acc1);
                acc1 = WMMA_BF16(ah, bl, acc1);
                acc1 = WMMA_BF16(al, bh, acc1);
            }
        }

        // running argmin: this lane's columns are codes c0 and c0+16
        const int   c0  = tileCode + cg * 32 + fi;
        const float en0 = enorm2[c0];
        const float en1 = enorm2[c0 + 16];
        #pragma unroll
        for (int v = 0; v < 8; ++v) {
            const float s0 = __builtin_fmaf(-2.0f, acc0[v], en0);
            const float s1 = __builtin_fmaf(-2.0f, acc1[v], en1);
            if (s0 < bestV[v]) { bestV[v] = s0; bestI[v] = c0; }
            if (s1 < bestV[v]) { bestV[v] = s1; bestI[v] = c0 + 16; }
        }

        wait_async0();      // next buffer's DMA complete (this wave)
        __syncthreads();    // all waves: done reading buf, next buffer visible
    }

    // ---- reduce across the 16 lanes holding each row (masks < 16: stays in half-wave)
    #pragma unroll
    for (int m = 1; m < 16; m <<= 1) {
        #pragma unroll
        for (int v = 0; v < 8; ++v) {
            const float ov = __shfl_xor(bestV[v], m, 32);
            const int   oi = __shfl_xor(bestI[v], m, 32);
            if (ov < bestV[v] || (ov == bestV[v] && oi < bestI[v])) {
                bestV[v] = ov; bestI[v] = oi;
            }
        }
    }
    if ((lane & 15) == 0) {
        #pragma unroll
        for (int v = 0; v < 8; ++v) {
            const int row = rg * 16 + v + ((lane >> 4) << 3);
            sMinV[cg * MTILE + row] = bestV[v];
            sMinI[cg * MTILE + row] = bestI[v];
        }
    }
    __syncthreads();

    // ---- combine the two code-groups; emit index
    if (tid < MTILE) {
        const float v0 = sMinV[tid], v1 = sMinV[MTILE + tid];
        const int   i0 = sMinI[tid], i1 = sMinI[MTILE + tid];
        const int best = (v1 < v0 || (v1 == v0 && i1 < i0)) ? i1 : i0;
        sIdx[tid] = best;
        outidx[rowBase + tid] = best;
    }
    __syncthreads();

    // ---- gather quantized = emb[idx] from exact fp32 codebook (coalesced float4)
    for (int i = tid; i < MTILE * (DIM / 4); i += 128) {
        const int r = i >> 6;
        const int c = (i & 63) << 2;
        const float4 v = *(const float4*)(emb + (size_t)sIdx[r] * DIM + c);
        *(float4*)(outq + (size_t)(rowBase + r) * DIM + c) = v;
    }
}

extern "C" void kernel_launch(void* const* d_in, const int* in_sizes, int n_in,
                              void* d_out, int out_size, void* d_ws, size_t ws_size,
                              hipStream_t stream) {
    const float* x   = (const float*)d_in[0];   // (16,1024,256) f32
    const float* emb = (const float*)d_in[1];   // (8192,256)    f32

    // workspace: e_hi (4MB) | e_lo (4MB) | enorm2 (32KB)  ~= 8.4 MB
    unsigned short* e_hi = (unsigned short*)d_ws;
    unsigned short* e_lo = e_hi + (size_t)KCODES * DIM;
    float* enorm2 = (float*)(e_lo + (size_t)KCODES * DIM);

    float* outq   = (float*)d_out;                                  // (16,1024,256) f32
    int*   outidx = (int*)((float*)d_out + (size_t)N_ROWS * DIM);   // (16,1024) i32

    prep_kernel<<<KCODES / 8, 256, 0, stream>>>(emb, e_hi, e_lo, enorm2);

    const size_t smemBytes =
        (size_t)(2 * XPLANE + 4 * EPLANE) * sizeof(unsigned short)
        + 2 * MTILE * sizeof(float) + 2 * MTILE * sizeof(int) + MTILE * sizeof(int);
    vq_argmin_kernel<<<N_ROWS / MTILE, 128, smemBytes, stream>>>(
        x, emb, e_hi, e_lo, enorm2, outq, outidx);
}
